// AAE_52819507806661
// MI455X (gfx1250) — matrix operand order repaired and checked by reading, not compiled, not run on hardware
//
#include <hip/hip_runtime.h>
#include <hip/hip_bf16.h>
#include <stdint.h>

typedef __attribute__((ext_vector_type(16))) __bf16       v16bf;
typedef __attribute__((ext_vector_type(8)))  float        v8f;
typedef __attribute__((ext_vector_type(8)))  unsigned int v8u;

#define BB 16384     // batch
#define VV 64        // vocab
#define DE 128       // embedding dim
#define HH 512       // hidden
#define BT 16        // batch rows per workgroup (one WMMA M-tile)
#define PAD_TOK 0
#define BOS_TOK 1
#define EOS_TOK 2

// ---------------------------------------------------------------------------
// WMMA helpers (CDNA5 gfx1250, wave32): D(16x16,f32) = A(16x32,bf16) x B(32x16,bf16) + C
// ---------------------------------------------------------------------------
__device__ __forceinline__ v8f wmma_bf16(v16bf a, v16bf b, v8f c) {
  return __builtin_amdgcn_wmma_f32_16x16x32_bf16(false, a, false, b, (short)0, c,
                                                 false, false);
}

// A-matrix 16x32 bf16, source row-major (lda elements). Layout (matches B):
// lanes 0-15: M=lane, K 0..15; lanes 16-31: same M, K 16..31.
__device__ __forceinline__ v16bf load_a_tile(const __bf16* tile, int lda, int lane) {
  const int m  = lane & 15;
  const int kh = (lane >> 4) << 4;                       // 0 or 16
  const __bf16* p = tile + m * lda + kh;
  uint4 lo = *reinterpret_cast<const uint4*>(p);         // K = kh .. kh+7
  uint4 hi = *reinterpret_cast<const uint4*>(p + 8);     // K = kh+8 .. kh+15
  v8u t = { lo.x, lo.y, lo.z, lo.w, hi.x, hi.y, hi.z, hi.w };
  return __builtin_bit_cast(v16bf, t);
}

// B-matrix 32x16 bf16 taken as Wᵀ, W stored row-major (N_total x ldk).
// lanes 0-15 carry K 0..15, lanes 16-31 carry K 16..31; VGPR v packs K=2v,2v+1.
__device__ __forceinline__ v16bf load_b_tile(const __bf16* W, int ldk, int nrow,
                                             int k0, int lane) {
  const int n  = lane & 15;
  const int kk = (lane >> 4) << 4;                       // 0 or 16
  const __bf16* p = W + (size_t)(nrow + n) * ldk + k0 + kk;
  uint4 lo = *reinterpret_cast<const uint4*>(p);         // K offs 0..7
  uint4 hi = *reinterpret_cast<const uint4*>(p + 8);     // K offs 8..15
  v8u t = { lo.x, lo.y, lo.z, lo.w, hi.x, hi.y, hi.z, hi.w };
  return __builtin_bit_cast(v16bf, t);
}

// ---------------------------------------------------------------------------
// Elementwise math
// ---------------------------------------------------------------------------
__device__ __forceinline__ float sigm(float x) { return 1.0f / (1.0f + __expf(-x)); }
__device__ __forceinline__ float tanh_(float x) {
  x = fminf(fmaxf(x, -15.0f), 15.0f);
  float e = __expf(2.0f * x);
  return (e - 1.0f) / (e + 1.0f);
}
__device__ __forceinline__ unsigned rng_hash(unsigned a, unsigned b, unsigned c) {
  unsigned x = a * 0x9E3779B1u ^ (b * 0x85EBCA77u + 0x165667B1u) ^ (c * 0xC2B2AE3Du);
  x ^= x >> 16; x *= 0x7FEB352Du; x ^= x >> 15; x *= 0x846CA68Bu; x ^= x >> 16;
  return x;
}

struct Gates { v8f i, f, g, o; };

// Accumulate the four gate tiles (cols ncol, +H, +2H, +3H) over ktiles K-tiles.
__device__ __forceinline__ void mm_gates(Gates& acc, const __bf16* A, int lda,
                                         int ktiles, const __bf16* W, int ldk,
                                         int ncol, int lane) {
  for (int kt = 0; kt < ktiles; ++kt) {
    v16bf a = load_a_tile(A + kt * 32, lda, lane);
    acc.i = wmma_bf16(a, load_b_tile(W, ldk, ncol,          kt * 32, lane), acc.i);
    acc.f = wmma_bf16(a, load_b_tile(W, ldk, ncol + HH,     kt * 32, lane), acc.f);
    acc.g = wmma_bf16(a, load_b_tile(W, ldk, ncol + 2 * HH, kt * 32, lane), acc.g);
    acc.o = wmma_bf16(a, load_b_tile(W, ldk, ncol + 3 * HH, kt * 32, lane), acc.o);
    if (kt + 1 < ktiles)  // hint the L2-resident weight stream forward
      __builtin_prefetch(W + (size_t)(ncol + (lane & 15)) * ldk + (kt + 1) * 32, 0, 0);
  }
}

// LSTM cell nonlinearities on one 16x16 tile; c stays in VGPRs.
__device__ __forceinline__ void cell_update(const Gates& acc, const float* bias,
                                            int ncol, int lane, v8f& cstate, v8f& hout) {
  const int n = ncol + (lane & 15);
  const float bi = bias[n], bff = bias[n + HH], bg = bias[n + 2 * HH], bo = bias[n + 3 * HH];
  #pragma unroll
  for (int r = 0; r < 8; ++r) {
    float iv = sigm(acc.i[r] + bi);
    float fv = sigm(acc.f[r] + bff);
    float gv = tanh_(acc.g[r] + bg);
    float ov = sigm(acc.o[r] + bo);
    float cn = fv * cstate[r] + iv * gv;
    cstate[r] = cn;
    hout[r]   = ov * tanh_(cn);
  }
}

__device__ __forceinline__ void store_h(__bf16* sh, const v8f& h, int ncol, int lane) {
  const int col = ncol + (lane & 15);
  const int mb  = (lane >> 4) * 8;
  #pragma unroll
  for (int r = 0; r < 8; ++r) sh[(mb + r) * HH + col] = (__bf16)h[r];
}

// ---------------------------------------------------------------------------
// Persistent decode kernel: 1024 blocks x 256 threads (8 waves), BT=16 rows/block.
// Wave w owns hidden columns [w*64, w*64+64). c lives in registers, h in LDS.
// ---------------------------------------------------------------------------
extern "C" __global__ void __launch_bounds__(256)
aae_lstm_decode_wmma(const __bf16* __restrict__ z_bf,  const __bf16* __restrict__ E_bf,
                     const __bf16* __restrict__ Wl2h,  const float*  __restrict__ b_l2h,
                     const __bf16* __restrict__ Wih0,  const __bf16* __restrict__ Whh0,
                     const float*  __restrict__ bias0,
                     const __bf16* __restrict__ Wih1,  const __bf16* __restrict__ Whh1,
                     const float*  __restrict__ bias1,
                     const __bf16* __restrict__ Wout,  const float*  __restrict__ b_out,
                     float* __restrict__ out, int T) {
  __shared__ __attribute__((aligned(16))) __bf16 sx [BT * DE];   //  4 KB
  __shared__ __attribute__((aligned(16))) __bf16 sh0[BT * HH];   // 16 KB
  __shared__ __attribute__((aligned(16))) __bf16 sh1[BT * HH];   // 16 KB
  __shared__ __attribute__((aligned(16))) float  slog[BT * VV];  //  4 KB
  __shared__ int sprev[BT];
  __shared__ int sdone[BT];

  const int tid  = threadIdx.x;
  const int lane = tid & 31;
  const int wave = tid >> 5;
  const int rowbase = blockIdx.x * BT;

  for (int i = tid; i < BT * HH; i += 256) { sh0[i] = (__bf16)0.0f; sh1[i] = (__bf16)0.0f; }
  if (tid < BT) { sprev[tid] = BOS_TOK; sdone[tid] = 0; }

  // c0 = z @ W_l2hᵀ + b_l2h ; replicated into both layers' cell state (registers).
  v8f c0r[4], c1r[4];
  #pragma unroll
  for (int t = 0; t < 4; ++t) {
    const int ncol = wave * 64 + t * 16;
    v8f acc = {};
    #pragma unroll
    for (int kt = 0; kt < DE / 32; ++kt) {
      v16bf a = load_a_tile(z_bf + (size_t)rowbase * DE + kt * 32, DE, lane);
      acc = wmma_bf16(a, load_b_tile(Wl2h, DE, ncol, kt * 32, lane), acc);
    }
    const float bl = b_l2h[ncol + (lane & 15)];
    #pragma unroll
    for (int r = 0; r < 8; ++r) acc[r] += bl;
    c0r[t] = acc;
    c1r[t] = acc;
  }
  __syncthreads();

  for (int step = 0; step < T; ++step) {
    // --- embedding gather: x = E[prev] -> LDS (256 threads x 16B) ---
    {
      const int r  = tid >> 4;
      const int c0 = (tid & 15) * 8;
      const int tok = sprev[r];
      *reinterpret_cast<uint4*>(&sx[r * DE + c0]) =
          *reinterpret_cast<const uint4*>(E_bf + (size_t)tok * DE + c0);
    }
    __syncthreads();

    v8f hs[4];

    // --- layer 0: g = x@Wih0ᵀ + h0@Whh0ᵀ + bias ---
    #pragma unroll
    for (int t = 0; t < 4; ++t) {
      const int ncol = wave * 64 + t * 16;
      Gates acc = {};
      mm_gates(acc, sx,  DE, DE / 32, Wih0, DE, ncol, lane);
      mm_gates(acc, sh0, HH, HH / 32, Whh0, HH, ncol, lane);
      cell_update(acc, bias0, ncol, lane, c0r[t], hs[t]);
    }
    __syncthreads();   // everyone done reading old h0
    #pragma unroll
    for (int t = 0; t < 4; ++t) store_h(sh0, hs[t], wave * 64 + t * 16, lane);
    __syncthreads();

    // --- layer 1: g = h0_new@Wih1ᵀ + h1_old@Whh1ᵀ + bias ---
    #pragma unroll
    for (int t = 0; t < 4; ++t) {
      const int ncol = wave * 64 + t * 16;
      Gates acc = {};
      mm_gates(acc, sh0, HH, HH / 32, Wih1, HH, ncol, lane);
      mm_gates(acc, sh1, HH, HH / 32, Whh1, HH, ncol, lane);
      cell_update(acc, bias1, ncol, lane, c1r[t], hs[t]);
    }
    __syncthreads();   // everyone done reading old h1
    #pragma unroll
    for (int t = 0; t < 4; ++t) store_h(sh1, hs[t], wave * 64 + t * 16, lane);
    __syncthreads();

    // --- logits = h1@Woutᵀ + b_out (4 N-tiles on waves 0..3) ---
    if (wave < 4) {
      v8f acc = {};
      for (int kt = 0; kt < HH / 32; ++kt) {
        v16bf a = load_a_tile(sh1 + kt * 32, HH, lane);
        acc = wmma_bf16(a, load_b_tile(Wout, HH, wave * 16, kt * 32, lane), acc);
      }
      const int n  = wave * 16 + (lane & 15);
      const float bo = b_out[n];
      const int mb = (lane >> 4) * 8;
      #pragma unroll
      for (int r = 0; r < 8; ++r) slog[(mb + r) * VV + n] = acc[r] + bo;
    }
    __syncthreads();

    // --- Gumbel-max categorical sample, EOS/PAD bookkeeping, emit token ---
    if (tid < BT) {
      const int row  = tid;
      const int grow = rowbase + row;
      float best = -3.0e38f; int bestv = 0;
      for (int v = 0; v < VV; ++v) {
        const unsigned h = rng_hash(0x2545F491u ^ (unsigned)grow, (unsigned)step, (unsigned)v);
        const float u   = (float)(h >> 8) * (1.0f / 16777216.0f) + (1.0f / 33554432.0f);
        const float gmb = -__logf(-__logf(u));
        const float val = slog[row * VV + v] + gmb;
        if (val > best) { best = val; bestv = v; }
      }
      const int e   = sdone[row] | (bestv == EOS_TOK);
      const int tok = e ? PAD_TOK : bestv;
      sdone[row] = e;
      sprev[row] = tok;
      out[(size_t)grow * T + step] = (float)tok;
    }
    __syncthreads();
  }
}

// ---------------------------------------------------------------------------
// Prep kernels: f32 -> bf16 weight/activation conversion, bias folding.
// ---------------------------------------------------------------------------
extern "C" __global__ void cvt_f32_to_bf16(const float* __restrict__ s,
                                           __bf16* __restrict__ d, int n) {
  int i = blockIdx.x * blockDim.x + threadIdx.x;
  if (i < n) d[i] = (__bf16)s[i];
}
extern "C" __global__ void add_bias_f32(const float* __restrict__ a,
                                        const float* __restrict__ b,
                                        float* __restrict__ d, int n) {
  int i = blockIdx.x * blockDim.x + threadIdx.x;
  if (i < n) d[i] = a[i] + b[i];
}

// ---------------------------------------------------------------------------
// Host entry
// ---------------------------------------------------------------------------
extern "C" void kernel_launch(void* const* d_in, const int* in_sizes, int n_in,
                              void* d_out, int out_size, void* d_ws, size_t ws_size,
                              hipStream_t stream) {
  (void)in_sizes; (void)n_in; (void)ws_size;
  const float* z     = (const float*)d_in[0];
  const float* E     = (const float*)d_in[1];
  const float* W_l2h = (const float*)d_in[2];
  const float* b_l2h = (const float*)d_in[3];
  const float* W_ih0 = (const float*)d_in[4];
  const float* W_hh0 = (const float*)d_in[5];
  const float* b_ih0 = (const float*)d_in[6];
  const float* b_hh0 = (const float*)d_in[7];
  const float* W_ih1 = (const float*)d_in[8];
  const float* W_hh1 = (const float*)d_in[9];
  const float* b_ih1 = (const float*)d_in[10];
  const float* b_hh1 = (const float*)d_in[11];
  const float* W_out = (const float*)d_in[12];
  const float* b_out = (const float*)d_in[13];
  // d_in[14] = max_len lives on device; T derived host-side from out_size.
  const int T = out_size / BB;

  char* base = (char*)d_ws;
  size_t off = 0;
  auto take = [&](size_t bytes) -> char* {
    char* p = base + off;
    off += (bytes + 255) & ~(size_t)255;
    return p;
  };
  __bf16* z_bf  = (__bf16*)take((size_t)BB * DE * 2);
  __bf16* E_bf  = (__bf16*)take((size_t)VV * DE * 2);
  __bf16* Wl2h  = (__bf16*)take((size_t)HH * DE * 2);
  __bf16* Wih0  = (__bf16*)take((size_t)4 * HH * DE * 2);
  __bf16* Whh0  = (__bf16*)take((size_t)4 * HH * HH * 2);
  __bf16* Wih1  = (__bf16*)take((size_t)4 * HH * HH * 2);
  __bf16* Whh1  = (__bf16*)take((size_t)4 * HH * HH * 2);
  __bf16* Woutb = (__bf16*)take((size_t)VV * HH * 2);
  float*  bias0 = (float*)take((size_t)4 * HH * 4);
  float*  bias1 = (float*)take((size_t)4 * HH * 4);

  auto cvt = [&](const float* s, __bf16* d, int n) {
    cvt_f32_to_bf16<<<dim3((n + 255) / 256), dim3(256), 0, stream>>>(s, d, n);
  };
  cvt(z,     z_bf,  BB * DE);
  cvt(E,     E_bf,  VV * DE);
  cvt(W_l2h, Wl2h,  HH * DE);
  cvt(W_ih0, Wih0,  4 * HH * DE);
  cvt(W_hh0, Whh0,  4 * HH * HH);
  cvt(W_ih1, Wih1,  4 * HH * HH);
  cvt(W_hh1, Whh1,  4 * HH * HH);
  cvt(W_out, Woutb, VV * HH);
  add_bias_f32<<<dim3((4 * HH + 255) / 256), dim3(256), 0, stream>>>(b_ih0, b_hh0, bias0, 4 * HH);
  add_bias_f32<<<dim3((4 * HH + 255) / 256), dim3(256), 0, stream>>>(b_ih1, b_hh1, bias1, 4 * HH);

  aae_lstm_decode_wmma<<<dim3(BB / BT), dim3(256), 0, stream>>>(
      z_bf, E_bf, Wl2h, b_l2h, Wih0, Whh0, bias0, Wih1, Whh1, bias1,
      Woutb, b_out, (float*)d_out, T);
}
